// KernelAxialMultiAttention_36429912605146
// MI455X (gfx1250) — matrix operand order repaired and checked by reading, not compiled, not run on hardware
//
#include <hip/hip_runtime.h>
#include <hip/hip_bf16.h>
#include <stdint.h>

// Problem constants (from reference)
#define B_   2
#define M_   64
#define T_   512
#define C_   768
#define NH   12
#define RTOT (B_ * M_ * T_)   // 65536 rows
#define EPSV 1e-6f

typedef __attribute__((ext_vector_type(16))) __bf16 v16bf;
typedef __attribute__((ext_vector_type(8)))  float  v8f;

#define XS_STRIDE 40   // 32 bf16 + pad (uint16 units; 80B rows keep 16B alignment)
#define WS_STRIDE 40

__device__ __forceinline__ unsigned short f2bf(float f) {
  unsigned u = __float_as_uint(f);
  unsigned r = u + 0x7FFFu + ((u >> 16) & 1u);   // round-to-nearest-even
  return (unsigned short)(r >> 16);
}

// ---- CDNA5 async global->LDS copy (ASYNCcnt) --------------------------------
__device__ __forceinline__ void async_copy_b128(unsigned lds_byte_off, const void* g) {
  asm volatile("global_load_async_to_lds_b128 %0, %1, off"
               :: "v"(lds_byte_off), "v"(g) : "memory");
}
__device__ __forceinline__ void wait_async0() {
  asm volatile("s_wait_asynccnt 0x0" ::: "memory");
}
__device__ __forceinline__ unsigned lds_off_of(const void* p) {
  return (unsigned)(uintptr_t)p;   // generic LDS addr: hw uses addr[31:0]
}

union ABFrag { v16bf v; unsigned u[8]; };

// A-matrix 16x32 bf16 fragment per CDNA5 layout:
// lanes 0-15 : row m, K=0..7 in v0..3 (2/VGPR), K=16..23 in v4..7
// lanes 16-31: row m, K=8..15 in v0..3,          K=24..31 in v4..7
__device__ __forceinline__ v16bf load_a(const unsigned short* xs, int row0, int lane) {
  int m = lane & 15, hi = lane >> 4;
  const unsigned short* p = xs + (row0 + m) * XS_STRIDE + hi * 8;
  ABFrag f;
#pragma unroll
  for (int j = 0; j < 4; ++j) {
    f.u[j]     = *(const unsigned*)(p + 2 * j);
    f.u[j + 4] = *(const unsigned*)(p + 16 + 2 * j);
  }
  return f.v;
}

// B-matrix 32x16 bf16 fragment: lane holds column n; lanes 0-15 K=0..7,16..23,
// lanes 16-31 K=8..15,24..31, 2 K-values per VGPR. ws holds W[o][c] rows.
__device__ __forceinline__ v16bf load_b(const unsigned short* ws, int col0, int lane) {
  int n = lane & 15, kh = lane >> 4;
  const unsigned short* p = ws + (col0 + n) * WS_STRIDE + kh * 8;
  ABFrag f;
#pragma unroll
  for (int j = 0; j < 8; ++j) f.u[j] = *(const unsigned*)(p + 2 * j + (j >> 2) * 8);
  return f.v;
}

__device__ __forceinline__ v8f wmma_bf16(v16bf a, v16bf b, v8f c) {
  return __builtin_amdgcn_wmma_f32_16x16x32_bf16(false, a, false, b, (short)0, c, false, false);
}

// One 32x32 wave tile GEMM step over a staged 128x32 x-tile and 64x32 w-tile.
__device__ __forceinline__ void gemm_step(const unsigned short* xsb,
                                          const unsigned short* wsb,
                                          int wr, int wc, int lane, v8f acc[2][2]) {
  v16bf a0 = load_a(xsb, wr * 32,      lane);
  v16bf a1 = load_a(xsb, wr * 32 + 16, lane);
  v16bf b0 = load_b(wsb, wc * 32,      lane);
  v16bf b1 = load_b(wsb, wc * 32 + 16, lane);
  acc[0][0] = wmma_bf16(a0, b0, acc[0][0]);
  acc[0][1] = wmma_bf16(a0, b1, acc[0][1]);
  acc[1][0] = wmma_bf16(a1, b0, acc[1][0]);
  acc[1][1] = wmma_bf16(a1, b1, acc[1][1]);
}
// Same, but two B operands sharing one A (q and v projections).
__device__ __forceinline__ void gemm_step2(const unsigned short* xsb,
                                           const unsigned short* wqb,
                                           const unsigned short* wvb,
                                           int wr, int wc, int lane,
                                           v8f qacc[2][2], v8f vacc[2][2]) {
  v16bf a0 = load_a(xsb, wr * 32,      lane);
  v16bf a1 = load_a(xsb, wr * 32 + 16, lane);
  {
    v16bf b0 = load_b(wqb, wc * 32,      lane);
    v16bf b1 = load_b(wqb, wc * 32 + 16, lane);
    qacc[0][0] = wmma_bf16(a0, b0, qacc[0][0]);
    qacc[0][1] = wmma_bf16(a0, b1, qacc[0][1]);
    qacc[1][0] = wmma_bf16(a1, b0, qacc[1][0]);
    qacc[1][1] = wmma_bf16(a1, b1, qacc[1][1]);
  }
  {
    v16bf b0 = load_b(wvb, wc * 32,      lane);
    v16bf b1 = load_b(wvb, wc * 32 + 16, lane);
    vacc[0][0] = wmma_bf16(a0, b0, vacc[0][0]);
    vacc[0][1] = wmma_bf16(a0, b1, vacc[0][1]);
    vacc[1][0] = wmma_bf16(a1, b0, vacc[1][0]);
    vacc[1][1] = wmma_bf16(a1, b1, vacc[1][1]);
  }
}

// Staged x chunk held in registers between global load and LDS store so the
// loadcnt wait can be sunk below the WMMA block of the previous chunk.
struct XStage { float4 f[4]; };

__device__ __forceinline__ XStage load_x_global(const float* __restrict__ xg,
                                                size_t rowbase, int kbase, int tid) {
  XStage s;
#pragma unroll
  for (int j = 0; j < 4; ++j) {
    int i = tid + j * 256;
    int r = i >> 3, c4 = (i & 7) << 2;
    s.f[j] = *(const float4*)(xg + (rowbase + (size_t)r) * C_ + kbase + c4);
  }
  return s;
}
__device__ __forceinline__ void store_x_lds(unsigned short* xs, const XStage& sd, int tid) {
#pragma unroll
  for (int j = 0; j < 4; ++j) {
    int i = tid + j * 256;
    int r = i >> 3, c4 = (i & 7) << 2;
    union { unsigned short s[4]; uint2 u; } p;
    p.s[0] = f2bf(sd.f[j].x); p.s[1] = f2bf(sd.f[j].y);
    p.s[2] = f2bf(sd.f[j].z); p.s[3] = f2bf(sd.f[j].w);
    *(uint2*)(xs + r * XS_STRIDE + c4) = p.u;
  }
}

// 64 out-features x 32 K of pre-converted bf16 weight, async DMA to LDS.
__device__ __forceinline__ void load_w_tile_async(const unsigned short* __restrict__ wbf,
                                                  unsigned lds_base,
                                                  int obase, int kbase, int tid) {
  int r = tid >> 2;            // 0..63
  int c = (tid & 3) * 8;       // bf16 elements, 16B granules
  async_copy_b128(lds_base + (unsigned)(r * WS_STRIDE + c) * 2,
                  wbf + (size_t)(obase + r) * C_ + kbase + c);
}

// butterfly sum across a 16-lane half (lane bits 0..3)
__device__ __forceinline__ float half16_sum(float v) {
  v += __shfl_xor(v, 1, 32);
  v += __shfl_xor(v, 2, 32);
  v += __shfl_xor(v, 4, 32);
  v += __shfl_xor(v, 8, 32);
  return v;
}

// ---------------------------------------------------------------------------
// Pass 0: convert Wq|Wk|Wv (fp32, 3*C*C) to bf16 in workspace.
// ---------------------------------------------------------------------------
__global__ void __launch_bounds__(256) wconv_kernel(
    const float* __restrict__ Wq, const float* __restrict__ Wk,
    const float* __restrict__ Wv, unsigned short* __restrict__ wbf) {
  const size_t WN = (size_t)C_ * C_;           // 589824, multiple of 1024
  size_t i4 = ((size_t)blockIdx.x * 256 + threadIdx.x) * 4;
  size_t which = i4 / WN;
  const float* src = (which == 0) ? Wq : (which == 1) ? Wk : Wv;
  float4 f = *(const float4*)(src + (i4 - which * WN));
  union { unsigned short s[4]; uint2 u; } p;
  p.s[0] = f2bf(f.x); p.s[1] = f2bf(f.y); p.s[2] = f2bf(f.z); p.s[3] = f2bf(f.w);
  *(uint2*)(wbf + i4) = p.u;
}

// ---------------------------------------------------------------------------
// Pass 1: k = elu(x@Wk.T+bk)+1, masked; reduce to ksum (atomic) and krs.
// Tile 128x64 (one head), 8 waves each 32x32. Double-buffered LDS.
// grid = (NH, row-tiles) so the 12 blocks sharing an x-slab are adjacent.
// ---------------------------------------------------------------------------
__global__ void __launch_bounds__(256) kpass_kernel(
    const float* __restrict__ x, const float* __restrict__ mask,
    const unsigned short* __restrict__ wkbf, const float* __restrict__ bk,
    float* __restrict__ ksum, float* __restrict__ krs) {
  __shared__ unsigned short xs[2][128 * XS_STRIDE];
  __shared__ unsigned short ws[2][64 * WS_STRIDE];
  __shared__ float rowred[128 * 2];
  __shared__ float colred[4 * 64];
  __shared__ float biasl[64];
  __shared__ float maskl[128];

  int tid  = threadIdx.x;
  int lane = tid & 31;
  int wave = tid >> 5;
  int wr = wave >> 1, wc = wave & 1;
  int head  = blockIdx.x;
  size_t rowbase = (size_t)blockIdx.y * 128;
  int obase = head * 64;
  unsigned wsb0 = lds_off_of(ws[0]), wsb1 = lds_off_of(ws[1]);
  if (tid < 64)  biasl[tid] = bk[obase + tid];
  if (tid < 128) maskl[tid] = mask[rowbase + tid];

  v8f acc[2][2] = {};
  // prologue: stage chunk 0 into buffer 0
  load_w_tile_async(wkbf, wsb0, obase, 0, tid);
  {
    XStage x0 = load_x_global(x, rowbase, 0, tid);
    store_x_lds(xs[0], x0, tid);
  }
  for (int kb = 0; kb < C_; kb += 64) {
    // ---- chunk kb in buffer 0; prefetch kb+32 into buffer 1 ----
    wait_async0();
    __syncthreads();
    load_w_tile_async(wkbf, wsb1, obase, kb + 32, tid);
    XStage xn = load_x_global(x, rowbase, kb + 32, tid);
    gemm_step(xs[0], ws[0], wr, wc, lane, acc);
    store_x_lds(xs[1], xn, tid);
    // ---- chunk kb+32 in buffer 1; prefetch kb+64 into buffer 0 ----
    wait_async0();
    __syncthreads();
    bool more = (kb + 64) < C_;
    if (more) {
      load_w_tile_async(wkbf, wsb0, obase, kb + 64, tid);
      xn = load_x_global(x, rowbase, kb + 64, tid);
    }
    gemm_step(xs[1], ws[1], wr, wc, lane, acc);
    if (more) store_x_lds(xs[0], xn, tid);
  }

  // ------ epilogue: elementwise k, then in-register reductions ------
  int n = lane & 15, hi = lane >> 4;
  float kv[2][2][8];
#pragma unroll
  for (int tr = 0; tr < 2; ++tr)
#pragma unroll
    for (int tc = 0; tc < 2; ++tc) {
      int col = wc * 32 + tc * 16 + n;
#pragma unroll
      for (int i = 0; i < 8; ++i) {
        int row = wr * 32 + tr * 16 + hi * 8 + i;   // D layout: M = i + (lane>>4)*8
        float v = acc[tr][tc][i] + biasl[col];
        v = (v > 0.f) ? (v + 1.f) : __expf(v);      // elu(v)+1
        kv[tr][tc][i] = v * maskl[row];
      }
    }
  // row sums (-> krs): butterfly over the 16 column-lanes
  float rs[2][8];
#pragma unroll
  for (int tr = 0; tr < 2; ++tr)
#pragma unroll
    for (int i = 0; i < 8; ++i)
      rs[tr][i] = half16_sum(kv[tr][0][i] + kv[tr][1][i]);
  if (n == 0) {
#pragma unroll
    for (int tr = 0; tr < 2; ++tr)
#pragma unroll
      for (int i = 0; i < 8; ++i)
        rowred[(wr * 32 + tr * 16 + hi * 8 + i) * 2 + wc] = rs[tr][i];
  }
  // col sums (-> ksum): combine the two hi halves
  float cs[2];
#pragma unroll
  for (int tc = 0; tc < 2; ++tc) {
    float s = 0.f;
#pragma unroll
    for (int tr = 0; tr < 2; ++tr)
#pragma unroll
      for (int i = 0; i < 8; ++i) s += kv[tr][tc][i];
    s += __shfl_xor(s, 16, 32);
    cs[tc] = s;
  }
  if (hi == 0) {
    colred[wr * 64 + wc * 32 + n]      = cs[0];
    colred[wr * 64 + wc * 32 + 16 + n] = cs[1];
  }
  __syncthreads();

  size_t bm  = rowbase >> 9;
  size_t bmn = bm * NH + head;
  int tloc   = (int)(rowbase & 511);
  if (tid < 64) {
    float s = colred[tid] + colred[64 + tid] + colred[128 + tid] + colred[192 + tid];
    atomicAdd(&ksum[bmn * 64 + tid], s);
  }
  if (tid < 128)
    krs[bmn * 512 + tloc + tid] = rowred[tid * 2] + rowred[tid * 2 + 1];
}

// ---------------------------------------------------------------------------
// Pass 2: q and v projections fused on one x tile.
//   q epilogue: invz[t] = 1/(q[t]·ksum + eps)      (row sums)
//   v epilogue: S[e]  += sum_t krs[t]*v[t,e]       (col sums, atomic)
// ---------------------------------------------------------------------------
__global__ void __launch_bounds__(256) qvpass_kernel(
    const float* __restrict__ x,
    const unsigned short* __restrict__ wqbf, const float* __restrict__ bq,
    const unsigned short* __restrict__ wvbf, const float* __restrict__ bv,
    const float* __restrict__ ksum, const float* __restrict__ krs,
    float* __restrict__ invz, float* __restrict__ Sbuf) {
  __shared__ unsigned short xs[2][128 * XS_STRIDE];
  __shared__ unsigned short wqs[2][64 * WS_STRIDE];
  __shared__ unsigned short wvs[2][64 * WS_STRIDE];
  __shared__ float rowred[128 * 2];
  __shared__ float colred[4 * 64];
  __shared__ float biasq[64], biasv[64], ksl[64];
  __shared__ float krsl[128];

  int tid  = threadIdx.x;
  int lane = tid & 31;
  int wave = tid >> 5;
  int wr = wave >> 1, wc = wave & 1;
  int head  = blockIdx.x;
  size_t rowbase = (size_t)blockIdx.y * 128;
  int obase = head * 64;
  size_t bm  = rowbase >> 9;
  size_t bmn = bm * NH + head;
  int tloc   = (int)(rowbase & 511);
  unsigned wqb0 = lds_off_of(wqs[0]), wqb1 = lds_off_of(wqs[1]);
  unsigned wvb0 = lds_off_of(wvs[0]), wvb1 = lds_off_of(wvs[1]);

  if (tid < 64) {
    biasq[tid] = bq[obase + tid];
    biasv[tid] = bv[obase + tid];
    ksl[tid]   = ksum[bmn * 64 + tid];
  }
  if (tid < 128) krsl[tid] = krs[bmn * 512 + tloc + tid];

  v8f qacc[2][2] = {}, vacc[2][2] = {};
  load_w_tile_async(wqbf, wqb0, obase, 0, tid);
  load_w_tile_async(wvbf, wvb0, obase, 0, tid);
  {
    XStage x0 = load_x_global(x, rowbase, 0, tid);
    store_x_lds(xs[0], x0, tid);
  }
  for (int kb = 0; kb < C_; kb += 64) {
    wait_async0();
    __syncthreads();
    load_w_tile_async(wqbf, wqb1, obase, kb + 32, tid);
    load_w_tile_async(wvbf, wvb1, obase, kb + 32, tid);
    XStage xn = load_x_global(x, rowbase, kb + 32, tid);
    gemm_step2(xs[0], wqs[0], wvs[0], wr, wc, lane, qacc, vacc);
    store_x_lds(xs[1], xn, tid);

    wait_async0();
    __syncthreads();
    bool more = (kb + 64) < C_;
    if (more) {
      load_w_tile_async(wqbf, wqb0, obase, kb + 64, tid);
      load_w_tile_async(wvbf, wvb0, obase, kb + 64, tid);
      xn = load_x_global(x, rowbase, kb + 64, tid);
    }
    gemm_step2(xs[1], wqs[1], wvs[1], wr, wc, lane, qacc, vacc);
    if (more) store_x_lds(xs[0], xn, tid);
  }

  int n = lane & 15, hi = lane >> 4;

  // ---- q: (elu(q)+1)*ksum[col], row-sum -> rowred ----
  {
    float rs[2][8];
#pragma unroll
    for (int tr = 0; tr < 2; ++tr)
#pragma unroll
      for (int i = 0; i < 8; ++i) rs[tr][i] = 0.f;
#pragma unroll
    for (int tc = 0; tc < 2; ++tc) {
      int col = wc * 32 + tc * 16 + n;
#pragma unroll
      for (int tr = 0; tr < 2; ++tr)
#pragma unroll
        for (int i = 0; i < 8; ++i) {
          float v = qacc[tr][tc][i] + biasq[col];
          v = (v > 0.f) ? (v + 1.f) : __expf(v);
          rs[tr][i] += v * ksl[col];
        }
    }
#pragma unroll
    for (int tr = 0; tr < 2; ++tr)
#pragma unroll
      for (int i = 0; i < 8; ++i) rs[tr][i] = half16_sum(rs[tr][i]);
    if (n == 0) {
#pragma unroll
      for (int tr = 0; tr < 2; ++tr)
#pragma unroll
        for (int i = 0; i < 8; ++i)
          rowred[(wr * 32 + tr * 16 + hi * 8 + i) * 2 + wc] = rs[tr][i];
    }
  }
  // ---- v: (v+bias)*krs[row], col-sum -> colred ----
  {
    float cs[2];
#pragma unroll
    for (int tc = 0; tc < 2; ++tc) {
      int col = wc * 32 + tc * 16 + n;
      float s = 0.f;
#pragma unroll
      for (int tr = 0; tr < 2; ++tr)
#pragma unroll
        for (int i = 0; i < 8; ++i) {
          int row = wr * 32 + tr * 16 + hi * 8 + i;
          s += (vacc[tr][tc][i] + biasv[col]) * krsl[row];
        }
      s += __shfl_xor(s, 16, 32);
      cs[tc] = s;
    }
    if (hi == 0) {
      colred[wr * 64 + wc * 32 + n]      = cs[0];
      colred[wr * 64 + wc * 32 + 16 + n] = cs[1];
    }
  }
  __syncthreads();
  if (tid < 128) {
    float s = rowred[tid * 2] + rowred[tid * 2 + 1];
    invz[bmn * 512 + tloc + tid] = 1.f / (s + EPSV);
  }
  if (tid < 64) {
    float s = colred[tid] + colred[64 + tid] + colred[128 + tid] + colred[192 + tid];
    atomicAdd(&Sbuf[bmn * 64 + tid], s);
  }
}

// ---------------------------------------------------------------------------
// Pass 3: P[bmn,c] = sum_e S[bmn,e] * Wp[c, n*64+e]
// ---------------------------------------------------------------------------
__global__ void __launch_bounds__(256) ppass_kernel(
    const float* __restrict__ Sbuf, const float* __restrict__ Wp,
    float* __restrict__ P) {
  __shared__ float Ss[64];
  int bmn = blockIdx.x;
  int nh  = bmn % NH;
  int c   = blockIdx.y * 256 + threadIdx.x;
  if (threadIdx.x < 64) Ss[threadIdx.x] = Sbuf[(size_t)bmn * 64 + threadIdx.x];
  __syncthreads();
  const float* wrow = Wp + (size_t)c * C_ + nh * 64;
  float s = 0.f;
#pragma unroll 8
  for (int e = 0; e < 64; ++e) s += Ss[e] * wrow[e];
  P[(size_t)bmn * C_ + c] = s;
}

// ---------------------------------------------------------------------------
// Pass 4: out[r,c] = bp[c] + sum_n invz[bmn,t] * P[bmn,c]   (rank-12 combine)
// ---------------------------------------------------------------------------
__global__ void __launch_bounds__(256) outpass_kernel(
    const float* __restrict__ invz, const float* __restrict__ P,
    const float* __restrict__ bp, float* __restrict__ out) {
  int r = blockIdx.x;
  int c = blockIdx.y * 256 + threadIdx.x;
  int bm = r >> 9, t = r & 511;
  float s = bp[c];
#pragma unroll
  for (int n = 0; n < NH; ++n) {
    size_t bmn = (size_t)bm * NH + n;
    s += invz[bmn * 512 + t] * P[bmn * C_ + c];
  }
  out[(size_t)r * C_ + c] = s;
}

extern "C" void kernel_launch(void* const* d_in, const int* in_sizes, int n_in,
                              void* d_out, int out_size, void* d_ws, size_t ws_size,
                              hipStream_t stream) {
  const float* x    = (const float*)d_in[0];
  const float* mask = (const float*)d_in[1];
  const float* Wq   = (const float*)d_in[2];
  const float* bq   = (const float*)d_in[3];
  const float* Wk   = (const float*)d_in[4];
  const float* bk   = (const float*)d_in[5];
  const float* Wv   = (const float*)d_in[6];
  const float* bv   = (const float*)d_in[7];
  const float* Wp   = (const float*)d_in[8];
  const float* bp   = (const float*)d_in[9];
  float* out = (float*)d_out;

  const size_t BMN = (size_t)B_ * M_ * NH;     // 1536 head groups
  float* ksum = (float*)d_ws;                  // BMN*64
  float* Sbuf = ksum + BMN * 64;               // BMN*64
  float* krs  = Sbuf + BMN * 64;               // BMN*512
  float* invz = krs  + BMN * 512;              // BMN*512
  float* P    = invz + BMN * 512;              // BMN*768
  unsigned short* wbf = (unsigned short*)(P + BMN * C_);  // 3*C*C bf16 (Wq|Wk|Wv)
  // total workspace use: ~15.3 MB

  // zero the atomically-accumulated buffers (ksum + Sbuf are contiguous)
  hipMemsetAsync(ksum, 0, BMN * 64 * 2 * sizeof(float), stream);

  dim3 blk(256);
  const size_t WN = (size_t)C_ * C_;
  wconv_kernel<<<dim3((unsigned)(3 * WN / 1024)), blk, 0, stream>>>(Wq, Wk, Wv, wbf);
  kpass_kernel<<<dim3(NH, RTOT / 128), blk, 0, stream>>>(
      x, mask, wbf + WN, bk, ksum, krs);
  qvpass_kernel<<<dim3(NH, RTOT / 128), blk, 0, stream>>>(
      x, wbf, bq, wbf + 2 * WN, bv, ksum, krs, invz, Sbuf);
  ppass_kernel<<<dim3((unsigned)BMN, 3), blk, 0, stream>>>(Sbuf, Wp, P);
  outpass_kernel<<<dim3(RTOT, 3), blk, 0, stream>>>(invz, P, bp, out);
}
